// Mamba2Block_4097398800775
// MI455X (gfx1250) — compile-verified, matrix operand order checked by reading, not run on hardware
//
#include <hip/hip_runtime.h>
#include <hip/hip_bf16.h>

// ---------------------------------------------------------------------------
// Problem constants (fixed by the reference harness)
// ---------------------------------------------------------------------------
#define BB    2
#define LSEQ  1024
#define TT    2048          // B * L tokens
#define DM    768
#define DI    1536
#define DXZ   3072          // 2 * DI
#define NST   64
#define DBCW  1664          // DI + 2*NST  (delta_pre | B | C)

typedef __bf16 bf16_t;
typedef bf16_t       v16bf __attribute__((ext_vector_type(16)));
typedef float        v8f   __attribute__((ext_vector_type(8)));
typedef unsigned int v4u   __attribute__((ext_vector_type(4)));

union Frag16 { v16bf v; v4u q[2]; unsigned int u[8]; };

__device__ __forceinline__ unsigned short f2bf(float f) {
    unsigned int u = __float_as_uint(f);
    unsigned int r = u + 0x7FFFu + ((u >> 16) & 1u);   // round-nearest-even
    return (unsigned short)(r >> 16);
}
__device__ __forceinline__ float bf2f(unsigned short b) {
    return __uint_as_float(((unsigned int)b) << 16);
}

// ---------------------------------------------------------------------------
// Weight pack: f32 (K x N row-major) -> bf16 WMMA-B fragment order.
// Packed element index = ((nt*Kt + kt)*32 + lane)*16 + i, where the (lane,i)
// element is  W[kt*32 + i + 8*(lane>>4) + (i>=8?8:0)][nt*16 + (lane&15)].
// GEMM then loads each B fragment as two coalesced b128 per lane.
// ---------------------------------------------------------------------------
__global__ void __launch_bounds__(256)
pack_b_kernel(const float* __restrict__ W, unsigned short* __restrict__ P,
              int K, int N)
{
    int idx = blockIdx.x * 256 + threadIdx.x;   // over K*N
    if (idx >= K * N) return;
    int i    = idx & 15;
    int lane = (idx >> 4) & 31;
    int tile = idx >> 9;
    int Kt   = K >> 5;
    int kt   = tile % Kt;
    int nt   = tile / Kt;
    int k = kt * 32 + i + 8 * (lane >> 4) + ((i >= 8) ? 8 : 0);
    int n = nt * 16 + (lane & 15);
    P[idx] = f2bf(W[(size_t)k * N + n]);
}

// ---------------------------------------------------------------------------
// LayerNorm over Dm per token -> bf16
// ---------------------------------------------------------------------------
__global__ void __launch_bounds__(256)
ln_kernel(const float* __restrict__ x, const float* __restrict__ g,
          const float* __restrict__ be, unsigned short* __restrict__ xn)
{
    const int t = blockIdx.x;
    const float* row = x + (size_t)t * DM;
    __shared__ float red[2][8];

    float s = 0.f, ss = 0.f;
    for (int i = threadIdx.x; i < DM; i += 256) {
        float v = row[i]; s += v; ss += v * v;
    }
    for (int off = 16; off > 0; off >>= 1) {
        s  += __shfl_down(s,  off, 32);
        ss += __shfl_down(ss, off, 32);
    }
    int lane = threadIdx.x & 31, w = threadIdx.x >> 5;
    if (lane == 0) { red[0][w] = s; red[1][w] = ss; }
    __syncthreads();
    if (threadIdx.x == 0) {
        float a = 0.f, b = 0.f;
        for (int i = 0; i < 8; ++i) { a += red[0][i]; b += red[1][i]; }
        red[0][0] = a; red[1][0] = b;
    }
    __syncthreads();
    float mean = red[0][0] * (1.0f / DM);
    float var  = red[1][0] * (1.0f / DM) - mean * mean;
    float inv  = rsqrtf(var + 1e-5f);
    for (int i = threadIdx.x; i < DM; i += 256) {
        float v = (row[i] - mean) * inv * g[i] + be[i];
        xn[(size_t)t * DM + i] = f2bf(v);
    }
}

// ---------------------------------------------------------------------------
// bf16 WMMA GEMM: C(MxN,f32) = A(MxK,bf16,row-major) * Bp(packed fragments)
// optional epilogue add (residual). One wave computes a 16x64 tile.
// grid = (N/64, M/128), block = 256 (8 waves). Dims: M%128==0, N%64==0, K%32==0.
// Inner loop: 2 b128 A-loads + 8 b128 B-loads + 4 WMMA.
// ---------------------------------------------------------------------------
__global__ void __launch_bounds__(256)
gemm_bf16_wmma(const unsigned short* __restrict__ A,
               const unsigned short* __restrict__ Bp,
               float* __restrict__ C,
               const float* __restrict__ addsrc,
               int M, int N, int K, int ldc)
{
    const int lane  = threadIdx.x & 31;
    const int wave  = threadIdx.x >> 5;
    const int mBase = (blockIdx.y * 8 + wave) * 16;
    const int nBase = blockIdx.x * 64;
    const int half  = lane >> 4;
    const int l15   = lane & 15;
    const int mRow  = mBase + l15;
    const int Kt    = K >> 5;
    const size_t tstride = (size_t)Kt * 512;   // packed elements per n-tile step

    v8f acc0 = {}, acc1 = {}, acc2 = {}, acc3 = {};

    // per-lane packed-B base for n-tile (nBase/16), k-tile 0
    const unsigned short* bbase =
        Bp + (((size_t)(nBase >> 4) * Kt) * 32 + lane) * 16;

    for (int k0 = 0; k0 < K; k0 += 32) {
        // ---- A fragment: two contiguous 16B runs per lane
        Frag16 a;
        const v4u* arow = (const v4u*)(A + (size_t)mRow * K + k0 + 8 * half);
        a.q[0] = arow[0];     // K = 8*half .. 8*half+7
        a.q[1] = arow[2];     // K = 16+8*half .. 16+8*half+7

        // ---- B fragments: 32 contiguous bytes per lane per tile
        const v4u* bp = (const v4u*)(bbase + (size_t)(k0 >> 5) * 512);
        Frag16 b0, b1, b2, b3;
        b0.q[0] = bp[0];                       b0.q[1] = bp[1];
        b1.q[0] = *(const v4u*)((const unsigned short*)bp + tstride);
        b1.q[1] = *(const v4u*)((const unsigned short*)bp + tstride + 8);
        b2.q[0] = *(const v4u*)((const unsigned short*)bp + 2 * tstride);
        b2.q[1] = *(const v4u*)((const unsigned short*)bp + 2 * tstride + 8);
        b3.q[0] = *(const v4u*)((const unsigned short*)bp + 3 * tstride);
        b3.q[1] = *(const v4u*)((const unsigned short*)bp + 3 * tstride + 8);

        acc0 = __builtin_amdgcn_wmma_f32_16x16x32_bf16(false, a.v, false, b0.v,
                                                       (short)0, acc0, false, false);
        acc1 = __builtin_amdgcn_wmma_f32_16x16x32_bf16(false, a.v, false, b1.v,
                                                       (short)0, acc1, false, false);
        acc2 = __builtin_amdgcn_wmma_f32_16x16x32_bf16(false, a.v, false, b2.v,
                                                       (short)0, acc2, false, false);
        acc3 = __builtin_amdgcn_wmma_f32_16x16x32_bf16(false, a.v, false, b3.v,
                                                       (short)0, acc3, false, false);
    }

    // ---- store: VGPR r holds M = mBase + r + 8*half, N = nBase + l15 (+tile*16)
    #pragma unroll
    for (int r = 0; r < 8; ++r) {
        int m = mBase + r + 8 * half;
        size_t o = (size_t)m * ldc + nBase + l15;
        if (addsrc) {
            C[o     ] = acc0[r] + addsrc[o     ];
            C[o + 16] = acc1[r] + addsrc[o + 16];
            C[o + 32] = acc2[r] + addsrc[o + 32];
            C[o + 48] = acc3[r] + addsrc[o + 48];
        } else {
            C[o     ] = acc0[r];
            C[o + 16] = acc1[r];
            C[o + 32] = acc2[r];
            C[o + 48] = acc3[r];
        }
    }
}

// ---------------------------------------------------------------------------
// Depthwise causal conv (K=4) + bias + SiLU on the x_main half of xz -> bf16
// ---------------------------------------------------------------------------
__global__ void __launch_bounds__(256)
conv_silu_kernel(const float* __restrict__ xz, const float* __restrict__ cw,
                 const float* __restrict__ cb, unsigned short* __restrict__ xmain)
{
    int idx = blockIdx.x * 256 + threadIdx.x;   // over TT*DI exactly
    int d = idx % DI;
    int r = idx / DI;          // global token row
    int t = r % LSEQ;          // position within sequence
    float acc = cb[d];
    #pragma unroll
    for (int j = 0; j < 4; ++j) {
        int tt = t - 3 + j;
        if (tt >= 0)
            acc += cw[d * 4 + j] * xz[(size_t)(r - 3 + j) * DXZ + d];
    }
    float s = acc / (1.f + __expf(-acc));       // SiLU
    xmain[(size_t)r * DI + d] = f2bf(s);
}

// ---------------------------------------------------------------------------
// Selective scan. One wave owns 32 channels (lane = d), 64 states per lane
// kept in VGPRs. B_t/C_t (512B, shared across d) staged into LDS each step
// with one global_load_async_to_lds_b128 (ASYNCcnt path, no VGPR round-trip).
// grid = 12, block = 256 (8 waves -> 96 wave-tasks = B * DI/32).
// ---------------------------------------------------------------------------
__global__ void __launch_bounds__(256)
scan_kernel(const float* __restrict__ dbc, const unsigned short* __restrict__ xmain,
            const float* __restrict__ xz, const float* __restrict__ b_dt,
            const float* __restrict__ A_log, const float* __restrict__ D_skip,
            unsigned short* __restrict__ ygate)
{
    const int lane = threadIdx.x & 31;
    const int w    = threadIdx.x >> 5;
    const int g    = blockIdx.x * 8 + w;        // 0..95
    const int nck  = DI / 32;                   // 48
    const int b    = g / nck;
    const int d    = (g % nck) * 32 + lane;

    __shared__ __align__(16) float sBC[8][128];     // [wave][ B(64) | C(64) ]
    // wave-relative LDS byte address for this wave's slice, + per-lane 16B
    const unsigned ldsAddr =
        (unsigned)(uintptr_t)(&sBC[w][0]) + (unsigned)lane * 16u;

    float h[NST];
    float nA[NST];
    #pragma unroll
    for (int n = 0; n < NST; ++n) {
        h[n]  = 0.f;
        nA[n] = -__expf(A_log[(size_t)d * NST + n]);
    }
    const float bdt = b_dt[d];
    const float dsk = D_skip[d];

    for (int t = 0; t < LSEQ; ++t) {
        size_t r = (size_t)b * LSEQ + t;
        const float* row = dbc + r * DBCW;

        // stage B_t|C_t (128 contiguous floats) straight into LDS
        unsigned long long gaddr =
            (unsigned long long)(uintptr_t)(row + DI) + (unsigned long long)lane * 16u;
        asm volatile("global_load_async_to_lds_b128 %0, %1, off"
                     :: "v"(ldsAddr), "v"(gaddr) : "memory");

        float dp    = row[d] + bdt;
        float delta = (dp > 20.f) ? dp : log1pf(__expf(dp));   // softplus
        float xt    = bf2f(xmain[r * DI + d]);
        float dx    = delta * xt;

        asm volatile("s_wait_asynccnt 0x0" ::: "memory");

        float y = 0.f;
        #pragma unroll
        for (int n = 0; n < NST; ++n) {
            float dA = __expf(delta * nA[n]);
            h[n] = dA * h[n] + dx * sBC[w][n];
            y += h[n] * sBC[w][64 + n];
        }
        y += xt * dsk;

        float z  = xz[r * DXZ + DI + d];
        float gz = z / (1.f + __expf(-z));      // SiLU gate
        ygate[r * DI + d] = f2bf(y * gz);
    }
}

// ---------------------------------------------------------------------------
// Host launch
// ---------------------------------------------------------------------------
extern "C" void kernel_launch(void* const* d_in, const int* in_sizes, int n_in,
                              void* d_out, int out_size, void* d_ws, size_t ws_size,
                              hipStream_t stream)
{
    const float* x      = (const float*)d_in[0];
    const float* ln_g   = (const float*)d_in[1];
    const float* ln_b   = (const float*)d_in[2];
    const float* W_in   = (const float*)d_in[3];
    const float* conv_w = (const float*)d_in[4];
    const float* conv_b = (const float*)d_in[5];
    const float* W_dt   = (const float*)d_in[6];
    const float* b_dt   = (const float*)d_in[7];
    const float* A_log  = (const float*)d_in[8];
    const float* D_skip = (const float*)d_in[9];
    const float* W_B    = (const float*)d_in[10];
    const float* W_C    = (const float*)d_in[11];
    const float* W_out  = (const float*)d_in[12];
    float* out = (float*)d_out;

    // ---- workspace layout (all sizes multiples of 256 bytes) ----
    char* ws = (char*)d_ws;
    size_t off = 0;
    unsigned short* xn_bf    = (unsigned short*)(ws + off); off += (size_t)TT * DM * 2;
    unsigned short* win_bf   = (unsigned short*)(ws + off); off += (size_t)DM * DXZ * 2;
    unsigned short* wdt_bf   = (unsigned short*)(ws + off); off += (size_t)DI * DI * 2;
    unsigned short* wb_bf    = (unsigned short*)(ws + off); off += (size_t)DI * NST * 2;
    unsigned short* wc_bf    = (unsigned short*)(ws + off); off += (size_t)DI * NST * 2;
    unsigned short* wout_bf  = (unsigned short*)(ws + off); off += (size_t)DI * DM * 2;
    float*          xz       = (float*)(ws + off);          off += (size_t)TT * DXZ * 4;
    unsigned short* xmain_bf = (unsigned short*)(ws + off); off += (size_t)TT * DI * 4; // pad
    float*          dbc      = (float*)(ws + off);          off += (size_t)TT * DBCW * 4;
    unsigned short* ygate_bf = (unsigned short*)(ws + off); off += (size_t)TT * DI * 2;

    // ---- weight pack (f32 -> bf16 WMMA fragment order) ----
    pack_b_kernel<<<(DM * DXZ) / 256, 256, 0, stream>>>(W_in,  win_bf,  DM, DXZ);
    pack_b_kernel<<<(DI * DI ) / 256, 256, 0, stream>>>(W_dt,  wdt_bf,  DI, DI);
    pack_b_kernel<<<(DI * NST) / 256, 256, 0, stream>>>(W_B,   wb_bf,   DI, NST);
    pack_b_kernel<<<(DI * NST) / 256, 256, 0, stream>>>(W_C,   wc_bf,   DI, NST);
    pack_b_kernel<<<(DI * DM ) / 256, 256, 0, stream>>>(W_out, wout_bf, DI, DM);

    // ---- LayerNorm ----
    ln_kernel<<<TT, 256, 0, stream>>>(x, ln_g, ln_b, xn_bf);

    // ---- in_proj: xz = xn @ W_in  (2048 x 3072 x 768) ----
    gemm_bf16_wmma<<<dim3(DXZ / 64, TT / 128), 256, 0, stream>>>(
        xn_bf, win_bf, xz, nullptr, TT, DXZ, DM, DXZ);

    // ---- depthwise causal conv + SiLU ----
    conv_silu_kernel<<<(TT * DI) / 256, 256, 0, stream>>>(xz, conv_w, conv_b, xmain_bf);

    // ---- dt/B/C projections into fused dbc buffer (ldc = 1664) ----
    gemm_bf16_wmma<<<dim3(DI / 64, TT / 128), 256, 0, stream>>>(
        xmain_bf, wdt_bf, dbc, nullptr, TT, DI, DI, DBCW);
    gemm_bf16_wmma<<<dim3(1, TT / 128), 256, 0, stream>>>(
        xmain_bf, wb_bf, dbc + DI, nullptr, TT, NST, DI, DBCW);
    gemm_bf16_wmma<<<dim3(1, TT / 128), 256, 0, stream>>>(
        xmain_bf, wc_bf, dbc + DI + NST, nullptr, TT, NST, DI, DBCW);

    // ---- selective scan + skip + gate ----
    scan_kernel<<<12, 256, 0, stream>>>(dbc, xmain_bf, xz, b_dt, A_log, D_skip, ygate_bf);

    // ---- out_proj with residual epilogue -> d_out ----
    gemm_bf16_wmma<<<dim3(DM / 64, TT / 128), 256, 0, stream>>>(
        ygate_bf, wout_bf, out, x, TT, DM, DI, DM);

    (void)in_sizes; (void)n_in; (void)out_size; (void)ws_size;
}
